// CrossAtt_16372415332398
// MI455X (gfx1250) — matrix operand order repaired and checked by reading, not compiled, not run on hardware
//
#include <hip/hip_runtime.h>

// CDNA5 / gfx1250 cross-attention:
//   q = query@Wq^T, k = x@Wk^T, v = x@Wv^T  (bf16 WMMA, f32 accum)
//   S = q@k^T ; rowmax (pre-mask, as in reference) ; masked exp ; softmax
//   out0 = P@v [B,Lq,D] f32 ; out1 = att_w [B,Lq,Lk] f32
//
// Matmuls: v_wmma_f32_16x16x32_bf16. Q tile staged to LDS with
// global_load_async_to_lds_b128 (ASYNCcnt); P kept as bf16 in LDS so PV
// A-fragments are plain ds_load_b128.

typedef __attribute__((ext_vector_type(16))) __bf16 v16bf;
typedef __attribute__((ext_vector_type(8)))  float  v8f;
typedef __attribute__((ext_vector_type(4)))  int    v4i_t;

#define B_   8
#define L_   2048
#define D_   128
#define SSTR 2052   // f32 score-row stride  (rows rotate 4 banks)
#define PSTR 2072   // bf16 P-row stride     (rows rotate 12 dwords -> conflict-free b128)
#define QSTR 136    // bf16 q-row stride     (rows rotate 4 banks)

union AF { v16bf v; uint4 u[2]; };
union PK8 { uint4 u; __bf16 e[8]; };

static __device__ __forceinline__ v8f wmma_bf16(const AF& a, const AF& b, v8f c) {
  return __builtin_amdgcn_wmma_f32_16x16x32_bf16(false, a.v, false, b.v,
                                                 (short)0, c, false, false);
}

// ---- async global->LDS helper (gfx1250 GLOBAL_LOAD_ASYNC_TO_LDS_B128) -----
// Builtin signature (from hipcc diagnostic): (int4 AS1* src, int4 AS3* dst,
// imm offset, imm cpol).
#if __has_builtin(__builtin_amdgcn_global_load_async_to_lds_b128)
#define HAVE_ASYNC_LDS 1
typedef __attribute__((address_space(1))) v4i_t* gas_ptr;
typedef __attribute__((address_space(3))) v4i_t* las_ptr;
static __device__ __forceinline__ void async_copy16(const void* gsrc, void* ldst) {
  gas_ptr g = (gas_ptr)(unsigned long long)(const char*)gsrc;       // generic == AS1 value
  las_ptr l = (las_ptr)(unsigned int)(unsigned long long)(char*)ldst; // low 32b = LDS offset
  __builtin_amdgcn_global_load_async_to_lds_b128(g, l, 0, 0);
}
static __device__ __forceinline__ void async_wait0() {
#if __has_builtin(__builtin_amdgcn_s_wait_asynccnt)
  __builtin_amdgcn_s_wait_asynccnt(0);
#else
  asm volatile("s_wait_asynccnt 0x0" ::: "memory");
#endif
}
#endif

// ---------------------------------------------------------------------------
// Kernel 1: Wq/Wk/Wv f32 -> bf16, row-major [out_ch][in_ch] (K-contiguous,
// i.e. exactly the WMMA B-fragment layout).
// ---------------------------------------------------------------------------
__global__ void cvt_weights(const float* __restrict__ wq,
                            const float* __restrict__ wk,
                            const float* __restrict__ wv,
                            __bf16* __restrict__ out) {
  int i = blockIdx.x * 256 + threadIdx.x;          // 3*128*128 = 49152 total
  float f = (i < 16384) ? wq[i] : (i < 32768) ? wk[i - 16384] : wv[i - 32768];
  out[i] = (__bf16)f;
}

// ---------------------------------------------------------------------------
// Kernel 2: projections. One wave = one 16x16 output tile, K=128 -> 4 WMMAs.
// mat 0/1 stored row-major bf16; mat 2 (V) stored transposed [b][d][l].
// ---------------------------------------------------------------------------
__global__ void project_qkv(const float* __restrict__ query,
                            const float* __restrict__ x,
                            const __bf16* __restrict__ wb,   // [3][128][128]
                            __bf16* __restrict__ qb,         // [B][L][D]
                            __bf16* __restrict__ kb,         // [B][L][D]
                            __bf16* __restrict__ vt) {       // [B][D][L]
  const int tid  = threadIdx.x;
  const int lane = tid & 31, l16 = lane & 15, h = lane >> 4;
  int gw = blockIdx.x * 8 + (tid >> 5);
  const int at  = gw & 7;   gw >>= 3;
  const int mt  = gw & 127; gw >>= 7;
  const int b   = gw & 7;   gw >>= 3;
  const int mat = gw;                   // 0=q 1=k 2=v

  const float* in = (mat == 0) ? query : x;
  const float*  irow = in + ((long)(b * L_ + mt * 16 + l16)) * D_;
  const __bf16* wrow = wb + (long)mat * D_ * D_ + (at * 16 + l16) * D_;

  v8f acc = {};
#pragma unroll
  for (int s = 0; s < 4; ++s) {
    AF a, bf;
    const float* p0 = irow + s * 32 + h * 8;
    const float* p1 = irow + s * 32 + 16 + h * 8;
#pragma unroll
    for (int i = 0; i < 8; ++i) { a.v[i] = (__bf16)p0[i]; a.v[8 + i] = (__bf16)p1[i]; }
    bf.u[0] = *(const uint4*)(const void*)(wrow + s * 32 + h * 8);
    bf.u[1] = *(const uint4*)(const void*)(wrow + s * 32 + 16 + h * 8);
    acc = wmma_bf16(a, bf, acc);
  }

  if (mat == 2) {            // vt[b][a][m]: 8 contiguous bf16 per lane
    PK8 pk;
#pragma unroll
    for (int r = 0; r < 8; ++r) pk.e[r] = (__bf16)acc[r];
    __bf16* dst = vt + ((long)(b * D_ + at * 16 + l16)) * L_ + mt * 16 + 8 * h;
    *(uint4*)(void*)dst = pk.u;
  } else {
    __bf16* buf = (mat == 0) ? qb : kb;
#pragma unroll
    for (int r = 0; r < 8; ++r) {
      int m = r + 8 * h;
      buf[((long)(b * L_ + mt * 16 + m)) * D_ + at * 16 + l16] = (__bf16)acc[r];
    }
  }
}

// ---------------------------------------------------------------------------
// Kernel 3: attention. Grid (128 q-tiles, 8 batches), block 256 = 8 waves.
// LDS: S f32 16x2052 | Pb bf16 16x2072 | Qt bf16 16x136 | reductions (~203 KB).
// ---------------------------------------------------------------------------
__global__ void attention(const __bf16* __restrict__ qb,
                          const __bf16* __restrict__ kb,
                          const __bf16* __restrict__ vt,
                          const int*    __restrict__ valid,
                          float* __restrict__ result,   // [B][Lq][D]
                          float* __restrict__ attw) {   // [B][Lq][Lk]
  extern __shared__ char smem[];
  float*  S      = (float*)smem;                                   // 131328 B
  __bf16* Pb     = (__bf16*)(smem + 16 * SSTR * 4);                //  66304 B
  __bf16* Qt     = (__bf16*)(smem + 16 * SSTR * 4 + 16 * PSTR * 2);//   4352 B
  float*  red    = (float*)(smem + 16 * SSTR * 4 + 16 * PSTR * 2 + 16 * QSTR * 2);
  float*  rowmax = red + 256;
  float*  rowinv = rowmax + 16;

  const int tid  = threadIdx.x;
  const int w    = tid >> 5, lane = tid & 31, l16 = lane & 15, h = lane >> 4;
  const int q0   = blockIdx.x * 16;
  const int b    = blockIdx.y;

  // ---- Stage the 16x128 bf16 q tile into LDS (async, 16 B per thread) ----
  {
    const int row = tid >> 4, chunk = tid & 15;
    const __bf16* gsrc = qb + ((long)(b * L_ + q0 + row)) * D_ + chunk * 8;
    __bf16*       ldst = Qt + row * QSTR + chunk * 8;
#ifdef HAVE_ASYNC_LDS
    async_copy16(gsrc, ldst);
    async_wait0();
#else
    *(uint4*)(void*)ldst = *(const uint4*)(const void*)gsrc;
#endif
  }
  __syncthreads();

  // ---- Phase 1: S = q @ k^T, waves split the 128 key n-tiles -------------
  const __bf16* qls = Qt + l16 * QSTR;                             // A row m=l16
  for (int jt = w; jt < 128; jt += 8) {
    const int j0 = jt * 16;
    const __bf16* kcol = kb + ((long)(b * L_ + j0 + l16)) * D_;    // B col j=l16
    if (jt + 8 < 128)   // prefetch next key tile (global_prefetch_b8)
      __builtin_prefetch(kb + ((long)(b * L_ + j0 + 128 + l16)) * D_, 0, 1);
    v8f acc = {};
#pragma unroll
    for (int s = 0; s < 4; ++s) {
      AF a, bf;
      a.u[0]  = *(const uint4*)(const void*)(qls + s * 32 + h * 8);
      a.u[1]  = *(const uint4*)(const void*)(qls + s * 32 + 16 + h * 8);
      bf.u[0] = *(const uint4*)(const void*)(kcol + s * 32 + h * 8);
      bf.u[1] = *(const uint4*)(const void*)(kcol + s * 32 + 16 + h * 8);
      acc = wmma_bf16(a, bf, acc);
    }
#pragma unroll
    for (int r = 0; r < 8; ++r)          // C layout: M = r + 8h, N = l16
      S[(r + 8 * h) * SSTR + j0 + l16] = acc[r];
  }
  __syncthreads();

  // ---- Phase 2: row max over *all* scores (reference maxes before mask) --
  const int r = tid >> 4, sub = tid & 15;
  float mx = -INFINITY;
  for (int c = sub; c < L_; c += 16) mx = fmaxf(mx, S[r * SSTR + c]);
  red[r * 16 + sub] = mx;
  __syncthreads();
  if (tid < 16) {
    float m = red[tid * 16];
#pragma unroll
    for (int i = 1; i < 16; ++i) m = fmaxf(m, red[tid * 16 + i]);
    rowmax[tid] = m;
  }
  __syncthreads();

  // ---- Phase 3: masked exp, row sum; keep E as f32 (att_w) + bf16 (PV) ---
  const float rm  = rowmax[r];
  const int* vrow = valid + ((long)(b * L_ + q0 + r)) * (long)L_;
  float lsum = 0.f;
  for (int c = sub; c < L_; c += 16) {
    float e = (vrow[c] <= 0) ? 0.f : __expf(S[r * SSTR + c] - rm);
    S[r * SSTR + c]  = e;
    Pb[r * PSTR + c] = (__bf16)e;
    lsum += e;
  }
  red[r * 16 + sub] = lsum;
  __syncthreads();
  if (tid < 16) {
    float s = 0.f;
#pragma unroll
    for (int i = 0; i < 16; ++i) s += red[tid * 16 + i];
    rowinv[tid] = (s > 0.f) ? 1.f / s : 0.f;   // all-masked row -> zeros (NaN->0)
  }
  __syncthreads();
  const float inv = rowinv[r];
  float* arow = attw + ((long)(b * L_ + q0 + r)) * (long)L_;
  for (int c = sub; c < L_; c += 16) arow[c] = S[r * SSTR + c] * inv;

  // ---- Phase 4: out = (E @ V) * inv ; wave w owns output dims [16w,16w+16)
  const int d0 = w * 16;
  const __bf16* vcol = vt + ((long)(b * D_ + d0 + l16)) * L_;      // B col d=l16
  const __bf16* prow = Pb + l16 * PSTR;                            // A row m=l16
  v8f acc = {};
  for (int s = 0; s < 64; ++s) {
    AF a, bf;
    const int c0 = s * 32;
    a.u[0]  = *(const uint4*)(const void*)(prow + c0 + h * 8);
    a.u[1]  = *(const uint4*)(const void*)(prow + c0 + 16 + h * 8);
    bf.u[0] = *(const uint4*)(const void*)(vcol + c0 + h * 8);
    bf.u[1] = *(const uint4*)(const void*)(vcol + c0 + 16 + h * 8);
    acc = wmma_bf16(a, bf, acc);
  }
  float* orow = result + ((long)(b * L_ + q0)) * D_;
#pragma unroll
  for (int rr = 0; rr < 8; ++rr) {
    int m = rr + 8 * h;
    orow[m * D_ + d0 + l16] = acc[rr] * rowinv[m];
  }
}

// ---------------------------------------------------------------------------
extern "C" void kernel_launch(void* const* d_in, const int* in_sizes, int n_in,
                              void* d_out, int out_size, void* d_ws, size_t ws_size,
                              hipStream_t stream) {
  const float* query = (const float*)d_in[0];  // [8,2048,128]
  const float* x     = (const float*)d_in[1];  // [8,2048,128]
  const int*   valid = (const int*)  d_in[2];  // [8,2048,2048]
  const float* Wq    = (const float*)d_in[3];  // [128,128]
  const float* Wk    = (const float*)d_in[4];
  const float* Wv    = (const float*)d_in[5];

  float* result = (float*)d_out;                         // [8,2048,128]
  float* attw   = result + (long)B_ * L_ * D_;           // [8,2048,2048]

  // workspace layout (bytes): bf16 weights | qb | kb | vt  (~12.1 MB)
  char* ws = (char*)d_ws;
  __bf16* wb = (__bf16*)(ws);                            // 96 KB
  __bf16* qb = (__bf16*)(ws + 98304);                    // 4 MB
  __bf16* kb = (__bf16*)(ws + 98304 + 4194304);          // 4 MB
  __bf16* vt = (__bf16*)(ws + 98304 + 2 * 4194304);      // 4 MB

  cvt_weights<<<192, 256, 0, stream>>>(Wq, Wk, Wv, wb);
  project_qkv<<<3072, 256, 0, stream>>>(query, x, wb, qb, kb, vt);

  const size_t lds = 16 * SSTR * 4 + 16 * PSTR * 2 + 16 * QSTR * 2 +
                     (256 + 16 + 16) * sizeof(float);    // ~203 KB
  attention<<<dim3(128, B_), 256, lds, stream>>>(qb, kb, vt, valid, result, attw);
}
